// EdgeTypeAwareGATLayer_48172353192356
// MI455X (gfx1250) — matrix operand order repaired and checked by reading, not compiled, not run on hardware
//
#include <hip/hip_runtime.h>
#include <hip/hip_bf16.h>
#include <stdint.h>

#define N_NODES 50000
#define E_EDGES 320000
#define T_TYPES 4
#define H_HEADS 4
#define D_INF   128
#define D_OUTF  128
#define HD      512            // H_HEADS * D_OUTF
#define NEG_SLOPE 0.2f
#define LN_EPS 1e-5f

typedef __attribute__((ext_vector_type(16))) __bf16 v16bf;
typedef __attribute__((ext_vector_type(16))) unsigned short v16u;
typedef __attribute__((ext_vector_type(8)))  float v8f;

// ---------- helpers ----------
__device__ __forceinline__ unsigned short f2bf(float f) {
  unsigned u = __float_as_uint(f);
  unsigned r = (u + 0x7fffu + ((u >> 16) & 1u)) >> 16;   // RNE
  return (unsigned short)r;
}
__device__ __forceinline__ float bf2f(unsigned short s) {
  return __uint_as_float(((unsigned)s) << 16);
}
__device__ __forceinline__ float leaky(float f) { return f >= 0.f ? f : NEG_SLOPE * f; }
// monotonic float<->uint mapping for atomicMax on floats
__device__ __forceinline__ unsigned encf(float f) {
  unsigned u = __float_as_uint(f);
  return (u & 0x80000000u) ? ~u : (u | 0x80000000u);
}
__device__ __forceinline__ float decf(unsigned u) {
  unsigned v = (u & 0x80000000u) ? (u & 0x7fffffffu) : ~u;
  return __uint_as_float(v);
}

// Load A-operand fragment for one 16x32 K-slab (A row-major, lda bf16 elems).
// k order used here (contiguous, matched by B): lanes0-15 hold K kb+0..15;
// lanes16-31 hold K kb+16..31.
__device__ __forceinline__ v16bf load_a_frag(const unsigned short* __restrict__ A,
                                             int lda, int arow, int kb, int half) {
  union { uint4 q[2]; v16u v; } ua;
  const unsigned short* ap = A + (size_t)arow * lda + kb + half * 16;
  ua.q[0] = *(const uint4*)(ap);
  ua.q[1] = *(const uint4*)(ap + 8);
  return __builtin_bit_cast(v16bf, ua.v);
}

// Load B-operand fragment from COLUMN-MAJOR (transposed) weights Bt[col][k],
// ldbt = K stride. B k order used here (same as A): lanes0-15 -> K kb+0..15 at column l15;
// lanes16-31 -> K kb+16..31. All 16 values contiguous -> two b128 loads.
__device__ __forceinline__ v16bf load_bt_frag(const unsigned short* __restrict__ Bt,
                                              int ldbt, int bcol, int kb, int half) {
  union { uint4 q[2]; v16u v; } ub;
  const unsigned short* bp = Bt + (size_t)bcol * ldbt + kb + half * 16;
  ub.q[0] = *(const uint4*)(bp);
  ub.q[1] = *(const uint4*)(bp + 8);
  return __builtin_bit_cast(v16bf, ub.v);
}

// ---------- kernels ----------
__global__ void k_f32_to_bf16(const float* __restrict__ in, unsigned short* __restrict__ out, int n) {
  int i = blockIdx.x * blockDim.x + threadIdx.x;
  if (i < n) out[i] = f2bf(in[i]);
}

// W_gat [T][128][512] -> bf16 column-major per type: out[t][col][k]
__global__ void k_wgat_transpose(const float* __restrict__ in, unsigned short* __restrict__ out) {
  int i = blockIdx.x * blockDim.x + threadIdx.x;
  if (i >= T_TYPES * D_INF * HD) return;
  int t = i / (D_INF * HD), r = i % (D_INF * HD);
  int k = r / HD, c = r % HD;
  out[(size_t)t * D_INF * HD + (size_t)c * D_INF + k] = f2bf(in[i]);
}

// W_fuse [512][128] -> bf16 column-major: out[col][k]
__global__ void k_wfuse_transpose(const float* __restrict__ in, unsigned short* __restrict__ out) {
  int i = blockIdx.x * blockDim.x + threadIdx.x;
  if (i >= HD * D_OUTF) return;
  int k = i / D_OUTF, c = i % D_OUTF;
  out[(size_t)c * HD + k] = f2bf(in[i]);
}

// h_t = x @ W_gat[t]  (N x 128) @ (128 x 512), bf16 in/out, f32 accum.
// One block = 16 rows x 512 cols. 8 waves; each wave owns 4 column tiles and
// reuses its A fragment across 4 WMMAs per K-slab (16 WMMAs per wave total).
__global__ void k_gemm_h(const unsigned short* __restrict__ xb,
                         const unsigned short* __restrict__ Wt,   // [512][128] col-major
                         unsigned short* __restrict__ hb) {
  const int wave = threadIdx.x >> 5;
  const int lane = threadIdx.x & 31;
  const int half = lane >> 4, l15 = lane & 15;
  const int rowBase = blockIdx.x * 16;
  const int arow = rowBase + l15;
  v8f acc0 = {}, acc1 = {}, acc2 = {}, acc3 = {};
#pragma unroll
  for (int kb = 0; kb < D_INF; kb += 32) {
    v16bf av = load_a_frag(xb, D_INF, arow, kb, half);
    v16bf b0 = load_bt_frag(Wt, D_INF, (wave +  0) * 16 + l15, kb, half);
    v16bf b1 = load_bt_frag(Wt, D_INF, (wave +  8) * 16 + l15, kb, half);
    v16bf b2 = load_bt_frag(Wt, D_INF, (wave + 16) * 16 + l15, kb, half);
    v16bf b3 = load_bt_frag(Wt, D_INF, (wave + 24) * 16 + l15, kb, half);
    acc0 = __builtin_amdgcn_wmma_f32_16x16x32_bf16(false, av, false, b0, (short)0, acc0, false, false);
    acc1 = __builtin_amdgcn_wmma_f32_16x16x32_bf16(false, av, false, b1, (short)0, acc1, false, false);
    acc2 = __builtin_amdgcn_wmma_f32_16x16x32_bf16(false, av, false, b2, (short)0, acc2, false, false);
    acc3 = __builtin_amdgcn_wmma_f32_16x16x32_bf16(false, av, false, b3, (short)0, acc3, false, false);
  }
  const int rOff = half * 8;           // C/D: lanes16-31 hold M = r+8
  v8f accs[4] = {acc0, acc1, acc2, acc3};
#pragma unroll
  for (int j = 0; j < 4; ++j) {
    int col = (wave + 8 * j) * 16 + l15;
#pragma unroll
    for (int r = 0; r < 8; ++r)
      hb[(size_t)(rowBase + r + rOff) * HD + col] = f2bf(accs[j][r]);
  }
}

// alpha_s/alpha_d per (node, head)
__global__ void k_alpha(const unsigned short* __restrict__ hb,
                        const float* __restrict__ a_s, const float* __restrict__ a_d,
                        float* __restrict__ as, float* __restrict__ ad) {
  int i = blockIdx.x * blockDim.x + threadIdx.x;
  if (i >= N_NODES * H_HEADS) return;
  int n = i >> 2, hh = i & 3;
  const unsigned short* hp = hb + (size_t)n * HD + hh * D_OUTF;
  const float* asv = a_s + hh * D_OUTF;
  const float* adv = a_d + hh * D_OUTF;
  float ss = 0.f, sd = 0.f;
  for (int d = 0; d < D_OUTF; ++d) {
    float hv = bf2f(hp[d]);
    ss += hv * asv[d];
    sd += hv * adv[d];
  }
  as[i] = ss;
  ad[i] = sd;
}

// seed segment-max with the self-loop logit (always present, guarantees finite max)
__global__ void k_m_init(const float* __restrict__ as, const float* __restrict__ ad,
                         unsigned* __restrict__ menc) {
  int i = blockIdx.x * blockDim.x + threadIdx.x;
  if (i >= N_NODES * H_HEADS) return;
  menc[i] = encf(leaky(as[i] + ad[i]));
}

__global__ void k_edge_max(const int* __restrict__ eidx, const int* __restrict__ etype, int t,
                           const float* __restrict__ as, const float* __restrict__ ad,
                           unsigned* __restrict__ menc) {
  int e = blockIdx.x * blockDim.x + threadIdx.x;
  if (e >= E_EDGES || etype[e] != t) return;
  int s = eidx[e], d = eidx[E_EDGES + e];
#pragma unroll
  for (int hh = 0; hh < H_HEADS; ++hh) {
    float logit = leaky(as[s * 4 + hh] + ad[d * 4 + hh]);
    atomicMax(&menc[d * 4 + hh], encf(logit));
  }
}

// denom := exp(self - m)  (plain store; first writer each type iteration)
__global__ void k_denom_self(const float* __restrict__ as, const float* __restrict__ ad,
                             const unsigned* __restrict__ menc, float* __restrict__ denom) {
  int i = blockIdx.x * blockDim.x + threadIdx.x;
  if (i >= N_NODES * H_HEADS) return;
  denom[i] = __expf(leaky(as[i] + ad[i]) - decf(menc[i]));
}

__global__ void k_edge_sum(const int* __restrict__ eidx, const int* __restrict__ etype, int t,
                           const float* __restrict__ as, const float* __restrict__ ad,
                           const unsigned* __restrict__ menc, float* __restrict__ denom) {
  int e = blockIdx.x * blockDim.x + threadIdx.x;
  if (e >= E_EDGES || etype[e] != t) return;
  int s = eidx[e], d = eidx[E_EDGES + e];
#pragma unroll
  for (int hh = 0; hh < H_HEADS; ++hh) {
    float logit = leaky(as[s * 4 + hh] + ad[d * 4 + hh]);
    atomicAdd(&denom[d * 4 + hh], __expf(logit - decf(menc[d * 4 + hh])));
  }
}

// agg := alpha_self * h[n]   (initializes agg; full overwrite)
__global__ void k_self_agg(const unsigned short* __restrict__ hb,
                           const float* __restrict__ as, const float* __restrict__ ad,
                           const unsigned* __restrict__ menc, const float* __restrict__ denom,
                           float* __restrict__ agg) {
  int i = blockIdx.x * blockDim.x + threadIdx.x;
  if (i >= N_NODES * D_OUTF) return;
  int n = i >> 7, d = i & 127;
#pragma unroll
  for (int hh = 0; hh < H_HEADS; ++hh) {
    int nh = n * 4 + hh;
    float alpha = __expf(leaky(as[nh] + ad[nh]) - decf(menc[nh])) / denom[nh];
    agg[(size_t)nh * D_OUTF + d] = alpha * bf2f(hb[(size_t)n * HD + hh * D_OUTF + d]);
  }
}

__global__ void k_edge_agg(const int* __restrict__ eidx, const int* __restrict__ etype, int t,
                           const unsigned short* __restrict__ hb,
                           const float* __restrict__ as, const float* __restrict__ ad,
                           const unsigned* __restrict__ menc, const float* __restrict__ denom,
                           float* __restrict__ agg) {
  long long i = (long long)blockIdx.x * blockDim.x + threadIdx.x;
  if (i >= (long long)E_EDGES * D_OUTF) return;
  int e = (int)(i >> 7), d = (int)(i & 127);
  if (etype[e] != t) return;
  int s = eidx[e], dd = eidx[E_EDGES + e];
#pragma unroll
  for (int hh = 0; hh < H_HEADS; ++hh) {
    int nh = dd * 4 + hh;
    float logit = leaky(as[s * 4 + hh] + ad[nh]);
    float alpha = __expf(logit - decf(menc[nh])) / denom[nh];
    atomicAdd(&agg[(size_t)nh * D_OUTF + d],
              alpha * bf2f(hb[(size_t)s * HD + hh * D_OUTF + d]));
  }
}

// x_cat[:, t*128+d] = (head-mean(agg) + b_gat[t]) * softmax(etw)[t], stored bf16
__global__ void k_combine(const float* __restrict__ agg, const float* __restrict__ bg,
                          const float* __restrict__ etw, int t,
                          unsigned short* __restrict__ xcatb) {
  int i = blockIdx.x * blockDim.x + threadIdx.x;
  if (i >= N_NODES * D_OUTF) return;
  int n = i >> 7, d = i & 127;
  float s = 0.f;
#pragma unroll
  for (int hh = 0; hh < H_HEADS; ++hh) s += agg[(size_t)(n * 4 + hh) * D_OUTF + d];
  float mean = s * (1.f / H_HEADS);
  float e0 = etw[0], e1 = etw[1], e2 = etw[2], e3 = etw[3];
  float mx = fmaxf(fmaxf(e0, e1), fmaxf(e2, e3));
  float x0 = __expf(e0 - mx), x1 = __expf(e1 - mx), x2 = __expf(e2 - mx), x3 = __expf(e3 - mx);
  float inv = 1.f / (x0 + x1 + x2 + x3);
  float wt = (t == 0 ? x0 : t == 1 ? x1 : t == 2 ? x2 : x3) * inv;
  float val = (mean + bg[t * D_OUTF + d]) * wt;
  xcatb[(size_t)n * HD + t * D_OUTF + d] = f2bf(val);
}

// y = x_cat @ W_fuse + b_fuse; LayerNorm; ReLU  (16 rows x 128 cols per block, 8 WMMA waves)
__global__ void k_fuse_ln(const unsigned short* __restrict__ xcatb,
                          const unsigned short* __restrict__ Wft,   // [128][512] col-major
                          const float* __restrict__ bfuse,
                          const float* __restrict__ gamma, const float* __restrict__ beta,
                          float* __restrict__ out) {
  __shared__ float tile[16][132];
  __shared__ float mu_s[16], rs_s[16];
  const int wave = threadIdx.x >> 5;
  const int lane = threadIdx.x & 31;
  const int half = lane >> 4, l15 = lane & 15;
  const int rowBase = blockIdx.x * 16;
  const int arow = rowBase + l15;
  const int bcol = wave * 16 + l15;
  v8f c = {};
#pragma unroll
  for (int kb = 0; kb < HD; kb += 32) {
    v16bf av = load_a_frag(xcatb, HD, arow, kb, half);
    v16bf bv = load_bt_frag(Wft, HD, bcol, kb, half);
    c = __builtin_amdgcn_wmma_f32_16x16x32_bf16(false, av, false, bv, (short)0, c, false, false);
  }
  const int rOff = half * 8;
  const int col = wave * 16 + l15;
#pragma unroll
  for (int r = 0; r < 8; ++r) tile[r + rOff][col] = c[r] + bfuse[col];
  __syncthreads();
  if (threadIdx.x < 16) {
    float s = 0.f, s2 = 0.f;
    for (int j = 0; j < 128; ++j) { float v = tile[threadIdx.x][j]; s += v; s2 += v * v; }
    float mu = s * (1.f / 128.f);
    float var = s2 * (1.f / 128.f) - mu * mu;
    mu_s[threadIdx.x] = mu;
    rs_s[threadIdx.x] = rsqrtf(var + LN_EPS);
  }
  __syncthreads();
#pragma unroll
  for (int q = 0; q < 8; ++q) {
    int e = threadIdx.x * 8 + q;
    int row = e >> 7, cc = e & 127;
    float v = (tile[row][cc] - mu_s[row]) * rs_s[row] * gamma[cc] + beta[cc];
    out[(size_t)(rowBase + row) * D_OUTF + cc] = fmaxf(v, 0.f);
  }
}

// ---------- workspace layout (bytes) ----------
#define OFF_XB     ((size_t)0)                       // N*128 bf16       = 12,800,000
#define OFF_WGT    ((size_t)12800000)                // T*512*128 bf16   =    524,288
#define OFF_WFT    ((size_t)13324288)                // 128*512 bf16     =    131,072
#define OFF_XCATB  ((size_t)13455360)                // N*512 bf16       = 51,200,000
#define OFF_HB     ((size_t)64655360)                // N*512 bf16       = 51,200,000
#define OFF_AS     ((size_t)115855360)               // N*4 f32          =    800,000
#define OFF_AD     ((size_t)116655360)               // N*4 f32          =    800,000
#define OFF_MENC   ((size_t)117455360)               // N*4 u32          =    800,000
#define OFF_DENOM  ((size_t)118255360)               // N*4 f32          =    800,000
#define OFF_AGG    ((size_t)119055360)               // N*4*128 f32     = 102,400,000
// total ~221.5 MB

extern "C" void kernel_launch(void* const* d_in, const int* in_sizes, int n_in,
                              void* d_out, int out_size, void* d_ws, size_t ws_size,
                              hipStream_t stream) {
  const float* x    = (const float*)d_in[0];
  const int*   eidx = (const int*)d_in[1];
  const int*   etyp = (const int*)d_in[2];
  const float* Wg   = (const float*)d_in[3];
  const float* a_s  = (const float*)d_in[4];
  const float* a_d  = (const float*)d_in[5];
  const float* bg   = (const float*)d_in[6];
  const float* etw  = (const float*)d_in[7];
  const float* Wf   = (const float*)d_in[8];
  const float* bfu  = (const float*)d_in[9];
  const float* gam  = (const float*)d_in[10];
  const float* bet  = (const float*)d_in[11];
  float* out = (float*)d_out;

  char* ws = (char*)d_ws;
  unsigned short* xb    = (unsigned short*)(ws + OFF_XB);
  unsigned short* Wgt   = (unsigned short*)(ws + OFF_WGT);
  unsigned short* Wft   = (unsigned short*)(ws + OFF_WFT);
  unsigned short* xcatb = (unsigned short*)(ws + OFF_XCATB);
  unsigned short* hb    = (unsigned short*)(ws + OFF_HB);
  float*    as    = (float*)(ws + OFF_AS);
  float*    ad    = (float*)(ws + OFF_AD);
  unsigned* menc  = (unsigned*)(ws + OFF_MENC);
  float*    denom = (float*)(ws + OFF_DENOM);
  float*    agg   = (float*)(ws + OFF_AGG);

  const int B = 256;
  // conversions + weight transposes (column-major bf16 for contiguous B fragments)
  k_f32_to_bf16<<<(N_NODES * D_INF + B - 1) / B, B, 0, stream>>>(x, xb, N_NODES * D_INF);
  k_wgat_transpose<<<(T_TYPES * D_INF * HD + B - 1) / B, B, 0, stream>>>(Wg, Wgt);
  k_wfuse_transpose<<<(HD * D_OUTF + B - 1) / B, B, 0, stream>>>(Wf, Wft);

  const int nbNH = (N_NODES * H_HEADS + B - 1) / B;   // 782
  const int nbE  = (E_EDGES + B - 1) / B;             // 1250
  const int nbND = (N_NODES * D_OUTF + B - 1) / B;    // 25000
  const int nbED = (int)(((long long)E_EDGES * D_OUTF + B - 1) / B); // 160000

  for (int t = 0; t < T_TYPES; ++t) {
    k_gemm_h<<<N_NODES / 16, B, 0, stream>>>(xb, Wgt + (size_t)t * D_INF * HD, hb);
    k_alpha<<<nbNH, B, 0, stream>>>(hb, a_s + (size_t)t * H_HEADS * D_OUTF,
                                    a_d + (size_t)t * H_HEADS * D_OUTF, as, ad);
    k_m_init<<<nbNH, B, 0, stream>>>(as, ad, menc);
    k_edge_max<<<nbE, B, 0, stream>>>(eidx, etyp, t, as, ad, menc);
    k_denom_self<<<nbNH, B, 0, stream>>>(as, ad, menc, denom);
    k_edge_sum<<<nbE, B, 0, stream>>>(eidx, etyp, t, as, ad, menc, denom);
    k_self_agg<<<nbND, B, 0, stream>>>(hb, as, ad, menc, denom, agg);
    k_edge_agg<<<nbED, B, 0, stream>>>(eidx, etyp, t, hb, as, ad, menc, denom, agg);
    k_combine<<<nbND, B, 0, stream>>>(agg, bg, etw, t, xcatb);
  }

  k_fuse_ln<<<N_NODES / 16, B, 0, stream>>>(xcatb, Wft, bfu, gam, bet, out);
}
